// DPSNR_37967510897038
// MI455X (gfx1250) — compile-verified, matrix operand order checked by reading, not run on hardware
//
#include <hip/hip_runtime.h>
#include <hip/hip_bf16.h>
#include <math.h>

// ---------------- problem constants (match reference) ----------------
#define VOCAB 32000
#define DD    512      // hidden dim
#define DQ    256      // query dim
#define PP    32768    // pool slots per phase
#define KK    32       // top-k
#define STEPS 8
#define EPSV  0.01f
#define BB    2
#define SS    512
#define TOK   (BB*SS)  // 1024 tokens

typedef unsigned short u16;
typedef unsigned int   u32;

// ---------------- WMMA vector types ----------------
typedef __attribute__((ext_vector_type(16))) __bf16 v16bf;
typedef __attribute__((ext_vector_type(8)))  __bf16 v8bf;   // 16 bytes
typedef __attribute__((ext_vector_type(8)))  float  v8f;

union BF16x16 { v16bf v; v8bf h[2]; u16 u[16]; };

__device__ __forceinline__ u16 f2bf(float f) {              // RNE fp32 -> bf16
    u32 b = __float_as_uint(f);
    b += 0x7FFFu + ((b >> 16) & 1u);
    return (u16)(b >> 16);
}

#define WMMA_BF16(a, b, c) \
    __builtin_amdgcn_wmma_f32_16x16x32_bf16(false, (a), false, (b), (short)0, (c), false, false)

// Fragment from pre-converted bf16 row-major [rows, ld] (ld multiple of 8, kb multiple of 32).
// ISA 16-bit layout: lanes 0-15 row=base+l, K {kb..kb+7, kb+16..23}; lanes 16-31 K {+8, +24}.
// Exactly two contiguous 16-byte loads per lane -> global_load_b128, zero conversion VALU.
__device__ __forceinline__ v16bf load_frag_bf16(const u16* __restrict__ A, int ld,
                                                int base, int kb, int lane) {
    const u16* p = A + (size_t)(base + (lane & 15)) * ld + kb + ((lane >> 4) << 3);
    BF16x16 r;
    r.h[0] = *(const v8bf*)(p);
    r.h[1] = *(const v8bf*)(p + 16);
    return r.v;
}

// Shared GEMM core: A [M,K] bf16 row-major, Bt [N,K] bf16 row-major (pre-transposed),
// wave accumulates NT adjacent 16x16 tiles, reusing one A fragment per k-slice.
template<int NT, int KT>
__device__ __forceinline__ void gemm_core(const u16* __restrict__ Abf, int lda,
                                          const u16* __restrict__ Bt, int ldb,
                                          int row0, int wcol0, int lane, v8f acc[NT]) {
#pragma unroll
    for (int kb = 0; kb < KT * 32; kb += 32) {
        v16bf a = load_frag_bf16(Abf, lda, row0, kb, lane);
#pragma unroll
        for (int t = 0; t < NT; ++t) {
            v16bf b = load_frag_bf16(Bt, ldb, wcol0 + t * 16, kb, lane);
            acc[t] = WMMA_BF16(a, b, acc[t]);
        }
    }
}

// ================= one-time conversion kernels (off hot path) =================

// flat fp32 -> bf16, two elements per thread (packed store)
__global__ void k_cvt_flat(const float* __restrict__ in, u32* __restrict__ out, int n2) {
    int i = blockIdx.x * blockDim.x + threadIdx.x;
    if (i < n2) {
        float x = in[2 * i], y = in[2 * i + 1];
        out[i] = (u32)f2bf(x) | ((u32)f2bf(y) << 16);
    }
}

// [K,N] fp32 row-major -> [N,K] bf16 row-major (transpose so GEMM B-columns are contiguous)
__global__ void k_cvt_transpose(const float* __restrict__ in, u16* __restrict__ out,
                                int Kd, int N) {
    int n = blockIdx.x * blockDim.x + threadIdx.x;
    int k0 = blockIdx.y * 32;
    if (n < N) {
        for (int k = k0; k < k0 + 32; ++k)
            out[(size_t)n * Kd + k] = f2bf(in[(size_t)k * N + n]);
    }
}

// ================= 1. encode: hidden = emb[ids] + pos ; zero ACT state =================
__global__ void k_encode(const int* __restrict__ ids, const float* __restrict__ emb,
                         const float* __restrict__ pos, float* __restrict__ hid,
                         u16* __restrict__ hid_bf, float* __restrict__ acc_h,
                         float* __restrict__ cum, float* __restrict__ nupd) {
    int t = blockIdx.x;                 // 0..1023
    int s = t & (SS - 1);
    int tok = ids[t];
    const float* e = emb + (size_t)tok * DD;
    const float* q = pos + (size_t)s * DD;
    for (int d = threadIdx.x; d < DD; d += blockDim.x) {
        float v = e[d] + q[d];
        hid[(size_t)t * DD + d]    = v;
        hid_bf[(size_t)t * DD + d] = f2bf(v);
        acc_h[(size_t)t * DD + d]  = 0.f;
    }
    if (threadIdx.x == 0) { cum[t] = 0.f; nupd[t] = 0.f; }
}

// ================= 2a. halt + ACT accumulate (wave32 per token) =================
__global__ void k_halt_act(const float* __restrict__ hid, const float* __restrict__ w_halt,
                           const float* __restrict__ b_halt, float* __restrict__ acc_h,
                           float* __restrict__ cum, float* __restrict__ nupd) {
    int lane = threadIdx.x & 31;
    int wave = threadIdx.x >> 5;
    int t = blockIdx.x * (blockDim.x >> 5) + wave;
    const float* h = hid + (size_t)t * DD;
    float s = 0.f;
#pragma unroll 4
    for (int d = lane; d < DD; d += 32) s += h[d] * w_halt[d];
#pragma unroll
    for (int off = 16; off > 0; off >>= 1) s += __shfl_xor(s, off, 32);  // all lanes get sum
    float halt = 1.f / (1.f + __expf(-(s + b_halt[0])));
    float c = cum[t];
    float still = (c < 1.f - EPSV) ? 1.f : 0.f;
    float p = halt * still;
    float newc = c + p;
    bool exceeded = (newc > 1.f - EPSV) && (still > 0.f);
    float weight = exceeded ? (1.f - c) : p;
    if (lane == 0) {
        cum[t]  = exceeded ? 1.f : newc;
        nupd[t] = nupd[t] + still;
    }
    float* a = acc_h + (size_t)t * DD;
#pragma unroll 4
    for (int d = lane; d < DD; d += 32) a[d] += weight * h[d];
}

// ================= 2b. query_bf = bf16(hidden @ W_q)  (M=1024,N=256,K=512) =================
__global__ void k_gemm_query(const u16* __restrict__ hid_bf, const u16* __restrict__ Wq_t,
                             u16* __restrict__ query_bf) {
    int lane = threadIdx.x & 31, wave = threadIdx.x >> 5;
    int row0 = blockIdx.x * 16;
    int wcol0 = wave * 32;                       // 8 waves x 2 tiles = 256 cols
    v8f acc[2] = {};
    gemm_core<2, DD / 32>(hid_bf, DD, Wq_t, DD, row0, wcol0, lane, acc);
    int mh = (lane >> 4) * 8;
#pragma unroll
    for (int t = 0; t < 2; ++t) {
        int n = wcol0 + t * 16 + (lane & 15);
#pragma unroll
        for (int j = 0; j < 8; ++j)
            query_bf[(size_t)(row0 + mh + j) * DQ + n] = f2bf(acc[t][j]);
    }
}

// ====== 2c. fused scores -> top-32 -> softmax -> gather/weighted-sum -> retr_bf ======
// Block: 16 tokens, 256 threads (8 waves). Wave w owns key columns [chunk+16w, +16).
__global__ void k_score_topk_retrieve(const u16* __restrict__ query_bf,
                                      const u16* __restrict__ keys_bf,    // [P, DQ] bf16
                                      const float* __restrict__ values,   // [P, D] fp32
                                      u16* __restrict__ retr_bf) {
    __shared__ float s_sc[16][128];
    __shared__ float s_top[16][KK];
    __shared__ int   s_idx[16][KK];
    __shared__ float s_w[16][KK];

    int tid = threadIdx.x;
    int lane = tid & 31, wave = tid >> 5;
    int row0 = blockIdx.x * 16;

    for (int i = tid; i < 16 * KK; i += blockDim.x) {
        s_top[i >> 5][i & 31] = -1e30f;
        s_idx[i >> 5][i & 31] = 0;
    }

    // Hoist all 8 A-fragments (16 tokens x full DQ=256) for the whole P loop.
    v16bf afrag[8];
#pragma unroll
    for (int i = 0; i < 8; ++i) afrag[i] = load_frag_bf16(query_bf, DQ, row0, i * 32, lane);
    __syncthreads();

    for (int pbase = 0; pbase < PP; pbase += 128) {
        int col0 = pbase + wave * 16;
        if (pbase + 128 < PP)   // prefetch next key chunk (global_prefetch_b8)
            __builtin_prefetch(keys_bf + (size_t)(pbase + 128 + wave * 16 + (lane & 15)) * DQ, 0, 1);
        v8f acc = {};
#pragma unroll
        for (int i = 0; i < 8; ++i) {
            // keys_bf row-major [P,DQ] == Bᵀ: column n of B is row (col0+n), contiguous in K.
            v16bf b = load_frag_bf16(keys_bf, DQ, col0, i * 32, lane);
            acc = WMMA_BF16(afrag[i], b, acc);
        }
        int n = lane & 15, mh = (lane >> 4) * 8;
#pragma unroll
        for (int j = 0; j < 8; ++j) s_sc[mh + j][wave * 16 + n] = acc[j];
        __syncthreads();

        if (tid < 16) {                       // streaming top-32, 1 thread/token
            int m = tid;
            float mn = s_top[m][0]; int mnp = 0;
#pragma unroll
            for (int k = 1; k < KK; ++k) if (s_top[m][k] < mn) { mn = s_top[m][k]; mnp = k; }
            for (int c = 0; c < 128; ++c) {
                float v = s_sc[m][c];
                if (v > mn) {
                    s_top[m][mnp] = v; s_idx[m][mnp] = pbase + c;
                    mn = s_top[m][0]; mnp = 0;
#pragma unroll
                    for (int k = 1; k < KK; ++k) if (s_top[m][k] < mn) { mn = s_top[m][k]; mnp = k; }
                }
            }
        }
        __syncthreads();
    }

    if (tid < 16) {                           // softmax over 32 survivors
        int m = tid;
        float mx = -1e30f;
#pragma unroll
        for (int k = 0; k < KK; ++k) mx = fmaxf(mx, s_top[m][k]);
        float sum = 0.f;
#pragma unroll
        for (int k = 0; k < KK; ++k) { float e = __expf(s_top[m][k] - mx); s_w[m][k] = e; sum += e; }
        float inv = 1.f / sum;
#pragma unroll
        for (int k = 0; k < KK; ++k) s_w[m][k] *= inv;
    }
    __syncthreads();

    // gather + weighted sum in fp32, store bf16 (consumed only as WMMA A-operand)
    int d = tid * 2;
    for (int m = 0; m < 16; ++m) {
        float a0 = 0.f, a1 = 0.f;
#pragma unroll 4
        for (int k = 0; k < KK; ++k) {
            const float* vp = values + (size_t)s_idx[m][k] * DD + d;
            float w = s_w[m][k];
            a0 += w * vp[0];
            a1 += w * vp[1];
        }
        retr_bf[(size_t)(row0 + m) * DD + d]     = f2bf(a0);
        retr_bf[(size_t)(row0 + m) * DD + d + 1] = f2bf(a1);
    }
}

// ======= 2d. hidden' = hidden + tanh(concat(hidden,retrieved) @ W_int) =======
// M=1024, N=512, K=1024 (two K=512 passes). Wint_t: [512 N, 1024 K] bf16.
__global__ void k_gemm_integrate(const float* __restrict__ hid,
                                 const u16* __restrict__ hid_bf,
                                 const u16* __restrict__ retr_bf,
                                 const u16* __restrict__ Wint_t,
                                 float* __restrict__ hidN, u16* __restrict__ hidN_bf) {
    int lane = threadIdx.x & 31, wave = threadIdx.x >> 5;
    int row0 = blockIdx.x * 16;
    int wcol0 = wave * 64;                       // 8 waves x 4 tiles = 512 cols
    v8f acc[4] = {};
    gemm_core<4, DD / 32>(hid_bf,  DD, Wint_t,       2 * DD, row0, wcol0, lane, acc);
    gemm_core<4, DD / 32>(retr_bf, DD, Wint_t + DD,  2 * DD, row0, wcol0, lane, acc);
    int mh = (lane >> 4) * 8;
#pragma unroll
    for (int t = 0; t < 4; ++t) {
        int n = wcol0 + t * 16 + (lane & 15);
#pragma unroll
        for (int j = 0; j < 8; ++j) {
            size_t off = (size_t)(row0 + mh + j) * DD + n;
            float v = hid[off] + tanhf(acc[t][j]);
            hidN[off]    = v;
            hidN_bf[off] = f2bf(v);
        }
    }
}

// ================= 3. ACT finalize: final (in-place acc_h) + bf16 mirror + ponder =======
__global__ void k_finalize(const float* __restrict__ hid, float* __restrict__ acc_h,
                           u16* __restrict__ fin_bf, const float* __restrict__ cum,
                           const float* __restrict__ nupd, float* __restrict__ ponder) {
    int t = blockIdx.x;
    float rem = fminf(fmaxf(1.f - cum[t], 0.f), 1.f);
    for (int d = threadIdx.x; d < DD; d += blockDim.x) {
        size_t off = (size_t)t * DD + d;
        float v = acc_h[off] + rem * hid[off];
        acc_h[off]  = v;
        fin_bf[off] = f2bf(v);
    }
    if (threadIdx.x == 0) ponder[t] = nupd[t] + rem;
}

// ================= 4. logits = final @ W_dec  (M=1024,N=32000,K=512) =================
__global__ void k_gemm_decode(const u16* __restrict__ fin_bf, const u16* __restrict__ Wdec_t,
                              float* __restrict__ logits) {
    int lane = threadIdx.x & 31, wave = threadIdx.x >> 5;
    int row0 = blockIdx.x * 16;
    int wcol0 = blockIdx.y * 256 + wave * 32;    // 8 waves x 2 tiles = 256 cols/block
    v8f acc[2] = {};
    gemm_core<2, DD / 32>(fin_bf, DD, Wdec_t, DD, row0, wcol0, lane, acc);
    int mh = (lane >> 4) * 8;
#pragma unroll
    for (int t = 0; t < 2; ++t) {
        int n = wcol0 + t * 16 + (lane & 15);
#pragma unroll
        for (int j = 0; j < 8; ++j)
            logits[(size_t)(row0 + mh + j) * VOCAB + n] = acc[t][j];
    }
}

// ================= host orchestration =================
extern "C" void kernel_launch(void* const* d_in, const int* in_sizes, int n_in,
                              void* d_out, int out_size, void* d_ws, size_t ws_size,
                              hipStream_t stream) {
    (void)in_sizes; (void)n_in; (void)out_size; (void)ws_size;

    const int*   ids     = (const int*)  d_in[0];
    const float* emb     = (const float*)d_in[1];
    const float* pos     = (const float*)d_in[2];
    const float* W_q     = (const float*)d_in[3];
    const float* w_halt  = (const float*)d_in[4];
    const float* b_halt  = (const float*)d_in[5];
    const float* W_int   = (const float*)d_in[6];
    const float* W_dec   = (const float*)d_in[7];
    const float* pkeys   = (const float*)d_in[8];   // [3, P, DQ]
    const float* pvals   = (const float*)d_in[9];   // [3, P, D]

    float* out    = (float*)d_out;
    float* logits = out;                                  // [B,S,VOCAB]
    float* ponder = out + (size_t)TOK * VOCAB;            // [B,S]

    // ---------------- workspace layout ----------------
    char* ws = (char*)d_ws;
    float* hidA  = (float*)ws;                 ws += (size_t)TOK * DD * 4;
    float* hidB  = (float*)ws;                 ws += (size_t)TOK * DD * 4;
    float* acc_h = (float*)ws;                 ws += (size_t)TOK * DD * 4;
    float* cum   = (float*)ws;                 ws += TOK * 4;
    float* nupd  = (float*)ws;                 ws += TOK * 4;
    u16* hidAbf  = (u16*)ws;                   ws += (size_t)TOK * DD * 2;
    u16* hidBbf  = (u16*)ws;                   ws += (size_t)TOK * DD * 2;
    u16* querybf = (u16*)ws;                   ws += (size_t)TOK * DQ * 2;
    u16* retrbf  = (u16*)ws;                   ws += (size_t)TOK * DD * 2;
    u16* finbf   = (u16*)ws;                   ws += (size_t)TOK * DD * 2;
    u16* Wq_t    = (u16*)ws;                   ws += (size_t)DQ * DD * 2;       // [256,512]
    u16* Wint_t  = (u16*)ws;                   ws += (size_t)DD * 2 * DD * 2;   // [512,1024]
    u16* Wdec_t  = (u16*)ws;                   ws += (size_t)VOCAB * DD * 2;    // [32000,512]
    u16* keysbf  = (u16*)ws;                   ws += (size_t)3 * PP * DQ * 2;   // [3,P,DQ]

    static const int phases[STEPS] = {0, 0, 1, 1, 1, 1, 2, 2};

    // one-time operand conversions (pure functions of inputs -> deterministic)
    {
        int n2 = 3 * PP * DQ / 2;
        k_cvt_flat<<<(n2 + 255) / 256, 256, 0, stream>>>(pkeys, (u32*)keysbf, n2);
        k_cvt_transpose<<<dim3(1, DD / 32), 256, 0, stream>>>(W_q, Wq_t, DD, DQ);
        k_cvt_transpose<<<dim3(2, (2 * DD) / 32), 256, 0, stream>>>(W_int, Wint_t, 2 * DD, DD);
        k_cvt_transpose<<<dim3(125, DD / 32), 256, 0, stream>>>(W_dec, Wdec_t, DD, VOCAB);
    }

    k_encode<<<TOK, 256, 0, stream>>>(ids, emb, pos, hidA, hidAbf, acc_h, cum, nupd);

    float* hid  = hidA;  float* hidN  = hidB;
    u16*   hbf  = hidAbf; u16*  hbfN  = hidBbf;
    for (int t = 0; t < STEPS; ++t) {
        k_halt_act<<<TOK / 8, 256, 0, stream>>>(hid, w_halt, b_halt, acc_h, cum, nupd);
        k_gemm_query<<<TOK / 16, 256, 0, stream>>>(hbf, Wq_t, querybf);
        int ph = phases[t];
        k_score_topk_retrieve<<<TOK / 16, 256, 0, stream>>>(
            querybf, keysbf + (size_t)ph * PP * DQ, pvals + (size_t)ph * PP * DD, retrbf);
        k_gemm_integrate<<<TOK / 16, 256, 0, stream>>>(hid, hbf, retrbf, Wint_t, hidN, hbfN);
        float* tf = hid; hid = hidN; hidN = tf;
        u16*   tb = hbf; hbf = hbfN; hbfN = tb;
    }

    k_finalize<<<TOK, 256, 0, stream>>>(hid, acc_h, finbf, cum, nupd, ponder);
    k_gemm_decode<<<dim3(TOK / 16, VOCAB / 256), 256, 0, stream>>>(finbf, Wdec_t, logits);
}
